// MultiHeadAttention_73993696575684
// MI455X (gfx1250) — matrix operand order repaired and checked by reading, not compiled, not run on hardware
//
#include <hip/hip_runtime.h>
#include <stdint.h>

typedef __attribute__((ext_vector_type(16))) _Float16 v16h;
typedef __attribute__((ext_vector_type(8)))  float    v8f;

#define N_HEADS     8
#define Q_DIM       512
#define INPUT_DIM   512
#define EMBED_DIM   512
#define HEAD_DIM    64
#define BATCH       8
#define TOTAL_TOK   4096
#define KV_TOK      3584          // TOTAL_TOK - Q_DIM
#define NORM_SCALE  0.125f        // 1/sqrt(64)
#define P_LDS_LD    72            // padded LDS row stride (halves): 144B rows -> 16B aligned
                                  // chunks for async b128, 36-bank stride -> conflict-free ds reads

// ---------------------------------------------------------------- WMMA core
__device__ __forceinline__ v8f wmma_f16(v16h a, v16h b, v8f c) {
  // D = A(16x32 f16) * B(32x16 f16) + C(16x16 f32)
  return __builtin_amdgcn_wmma_f32_16x16x32_f16(false, a, false, b, (short)0, c, false, false);
}

// ------------------------------------------- CDNA5 async copy (ASYNCcnt path)
__device__ __forceinline__ uint32_t lds_off(const void* p) {
  // shared-aperture generic pointer: low 32 bits are the workgroup LDS offset
  return (uint32_t)(uintptr_t)p;
}
__device__ __forceinline__ void async_copy_b128(uint32_t dst_lds, const _Float16* src) {
  asm volatile("global_load_async_to_lds_b128 %0, %1, off"
               :: "v"(dst_lds), "v"(src) : "memory");
}
__device__ __forceinline__ void wait_async0() {
  asm volatile("s_wait_asynccnt 0x0" ::: "memory");
}

// A fragment (16x32). Lane L -> row M=L%16; halves e: K = k0 + 16*(L/16) + e
// (16 contiguous halves per lane, same k order as B).
__device__ __forceinline__ v16h load_a_f16(const _Float16* __restrict__ p, int ld, int row0, int k0) {
  int lane = threadIdx.x & 31;
  const _Float16* src = p + (size_t)(row0 + (lane & 15)) * ld + k0 + ((lane >> 4) << 4);
  v16h a;
  #pragma unroll
  for (int e = 0; e < 16; ++e) a[e] = src[e];
  return a;
}

__device__ __forceinline__ v16h load_a_f32(const float* __restrict__ p, int ld, int row0, int k0) {
  int lane = threadIdx.x & 31;
  const float* src = p + (size_t)(row0 + (lane & 15)) * ld + k0 + ((lane >> 4) << 4);
  v16h a;
  #pragma unroll
  for (int e = 0; e < 16; ++e) a[e] = (_Float16)src[e];
  return a;
}

// B fragment (32x16) from [N][K] storage. Lane L -> column N=n0+L%16,
// halves e: K = k0 + 16*(L/16) + e  (16 contiguous halves per lane).
__device__ __forceinline__ v16h load_b_f16(const _Float16* __restrict__ p, int ld, int n0, int k0) {
  int lane = threadIdx.x & 31;
  const _Float16* src = p + (size_t)(n0 + (lane & 15)) * ld + k0 + ((lane >> 4) << 4);
  v16h b;
  #pragma unroll
  for (int e = 0; e < 16; ++e) b[e] = src[e];
  return b;
}

__device__ __forceinline__ v16h load_b_f32(const float* __restrict__ p, int ld, int n0, int k0) {
  int lane = threadIdx.x & 31;
  const float* src = p + (size_t)(n0 + (lane & 15)) * ld + k0 + ((lane >> 4) << 4);
  v16h b;
  #pragma unroll
  for (int e = 0; e < 16; ++e) b[e] = (_Float16)src[e];
  return b;
}

// C/D fragment: lane L -> N = n0+L%16, VGPR j -> row M = row0 + j + 8*(L/16).
__device__ __forceinline__ void store_d_f16(_Float16* __restrict__ p, int ld, int row0, int n0, v8f d) {
  int lane = threadIdx.x & 31;
  int n = n0 + (lane & 15);
  int r = row0 + ((lane >> 4) << 3);
  #pragma unroll
  for (int j = 0; j < 8; ++j) p[(size_t)(r + j) * ld + n] = (_Float16)d[j];
}

__device__ __forceinline__ void store_d_f32(float* __restrict__ p, int ld, int row0, int n0, v8f d) {
  int lane = threadIdx.x & 31;
  int n = n0 + (lane & 15);
  int r = row0 + ((lane >> 4) << 3);
  #pragma unroll
  for (int j = 0; j < 8; ++j) p[(size_t)(r + j) * ld + n] = d[j];
}

// Row-wise reductions across one 16-lane half (xor masks 1..8 never cross halves).
__device__ __forceinline__ float red_max16(float v) {
  #pragma unroll
  for (int msk = 1; msk <= 8; msk <<= 1) v = fmaxf(v, __shfl_xor(v, msk, 32));
  return v;
}
__device__ __forceinline__ float red_add16(float v) {
  #pragma unroll
  for (int msk = 1; msk <= 8; msk <<= 1) v += __shfl_xor(v, msk, 32);
  return v;
}

// ------------------------------------------------ kernel 0: weight prep (f32 -> f16, transpose)
// wqT/wkT[h][k][i], wvT[h][v][i], woT[e][h*64+v]   (wqT/wkT/woT [N][K] for B loads, wvT [M][K] for A loads)
__global__ __launch_bounds__(256) void prep_kernel(
    const float* __restrict__ Wq, const float* __restrict__ Wk,
    const float* __restrict__ Wv, const float* __restrict__ Wo,
    _Float16* __restrict__ wqT, _Float16* __restrict__ wkT,
    _Float16* __restrict__ wvT, _Float16* __restrict__ woT) {
  int i = blockIdx.x * 256 + threadIdx.x;            // 0 .. 8*512*64-1
  int h  = i >> 15;
  int rm = i & 32767;
  int in = rm >> 6;                                  // input-dim index
  int k  = rm & 63;                                  // head-dim index
  size_t dst = (((size_t)h << 6) + k) * INPUT_DIM + in;
  wqT[dst] = (_Float16)Wq[i];
  wkT[dst] = (_Float16)Wk[i];
  wvT[dst] = (_Float16)Wv[i];
  int c = i >> 9, e = i & 511;                       // Wo[h][v][e] = Wo[c][e]
  woT[((size_t)e << 9) + c] = (_Float16)Wo[i];
}

// ------------------------------------------------ kernel 1/2: Q,K projection (pipelined K-loop)
__global__ __launch_bounds__(256) void qk_proj_kernel(
    const float* __restrict__ q, const _Float16* __restrict__ WT,
    _Float16* __restrict__ outF, int rowsPerHB, int rowOffset) {
  int wid = threadIdx.x >> 5;
  int hb  = blockIdx.y;                              // h*8 + b
  int h = hb >> 3, b = hb & 7;
  int r0 = blockIdx.x * 128 + wid * 16;
  const float*    A = q + ((size_t)b * TOTAL_TOK + rowOffset) * INPUT_DIM;
  const _Float16* B = WT + ((size_t)h << 6) * INPUT_DIM;   // [k][i], ld 512
  v8f acc[4] = {};
  v16h a = load_a_f32(A, INPUT_DIM, r0, 0);
  v16h bb[4];
  #pragma unroll
  for (int t = 0; t < 4; ++t) bb[t] = load_b_f16(B, INPUT_DIM, t * 16, 0);
  for (int k0 = 0; k0 < INPUT_DIM; k0 += 32) {
    int kn = (k0 + 32) & (INPUT_DIM - 1);            // wrap on last iter (values unused)
    v16h a_n = load_a_f32(A, INPUT_DIM, r0, kn);
    v16h bb_n[4];
    #pragma unroll
    for (int t = 0; t < 4; ++t) bb_n[t] = load_b_f16(B, INPUT_DIM, t * 16, kn);
    #pragma unroll
    for (int t = 0; t < 4; ++t) acc[t] = wmma_f16(a, bb[t], acc[t]);
    a = a_n;
    #pragma unroll
    for (int t = 0; t < 4; ++t) bb[t] = bb_n[t];
  }
  _Float16* O = outF + (size_t)hb * rowsPerHB * HEAD_DIM;
  #pragma unroll
  for (int t = 0; t < 4; ++t) store_d_f16(O, HEAD_DIM, r0, t * 16, acc[t]);
}

// ------------------------------------------------ kernel 3: V projection, produced TRANSPOSED
// Vt[hb][v][g] = sum_i Wv[h][i][v] * hrow[g][i];  M=v(64), N=g(3584), K=i(512)
__global__ __launch_bounds__(256) void v_proj_kernel(
    const float* __restrict__ q, const _Float16* __restrict__ WvT,
    _Float16* __restrict__ Vt) {
  int wid = threadIdx.x >> 5;
  int hb  = blockIdx.y;
  int h = hb >> 3, b = hb & 7;
  int t  = blockIdx.x * 8 + wid;                     // 0..223
  int m0 = (t & 3) * 16;                             // v tile
  int n0 = (t >> 2) * 64;                            // g block
  const _Float16* A  = WvT + ((size_t)h << 6) * INPUT_DIM;              // [v][i]
  const float*    Bs = q + ((size_t)b * TOTAL_TOK + Q_DIM) * INPUT_DIM; // h rows [g][i]
  v8f acc[4] = {};
  v16h a = load_a_f16(A, INPUT_DIM, m0, 0);
  v16h bb[4];
  #pragma unroll
  for (int tt = 0; tt < 4; ++tt) bb[tt] = load_b_f32(Bs, INPUT_DIM, n0 + tt * 16, 0);
  for (int k0 = 0; k0 < INPUT_DIM; k0 += 32) {
    int kn = (k0 + 32) & (INPUT_DIM - 1);
    v16h a_n = load_a_f16(A, INPUT_DIM, m0, kn);
    v16h bb_n[4];
    #pragma unroll
    for (int tt = 0; tt < 4; ++tt) bb_n[tt] = load_b_f32(Bs, INPUT_DIM, n0 + tt * 16, kn);
    #pragma unroll
    for (int tt = 0; tt < 4; ++tt) acc[tt] = wmma_f16(a, bb[tt], acc[tt]);
    a = a_n;
    #pragma unroll
    for (int tt = 0; tt < 4; ++tt) bb[tt] = bb_n[tt];
  }
  _Float16* O = Vt + (size_t)hb * HEAD_DIM * KV_TOK;
  #pragma unroll
  for (int tt = 0; tt < 4; ++tt) store_d_f16(O, KV_TOK, m0, n0 + tt * 16, acc[tt]);
}

// ------------------------------------------------ kernel 4: flash attention
// K/V g-blocks staged in LDS via async-to-LDS copies, double buffered across the
// whole workgroup (8 waves share each block: 8x fewer global requests, copy of
// block i+1 overlaps compute of block i).  Online softmax as before.
__global__ __launch_bounds__(256) void attn_kernel(
    const _Float16* __restrict__ Qf, const _Float16* __restrict__ Kf,
    const _Float16* __restrict__ Vt, _Float16* __restrict__ Hf) {
  __shared__ _Float16 Kbuf[2][64][P_LDS_LD];         // [g_local][k]  == B's [N][K]
  __shared__ _Float16 Vbuf[2][64][P_LDS_LD];         // [v][g_local]  == B's [N][K]
  __shared__ _Float16 Plds[8][16 * P_LDS_LD];        // per-wave P transpose staging

  int tid  = threadIdx.x;
  int wid  = tid >> 5;
  int lane = tid & 31;
  int hb = blockIdx.y;
  int h = hb >> 3, b = hb & 7;
  int q0 = blockIdx.x * 128 + wid * 16;

  const _Float16* Qhb = Qf + (size_t)hb * Q_DIM   * HEAD_DIM;
  const _Float16* Khb = Kf + (size_t)hb * KV_TOK  * HEAD_DIM;   // [g][k]
  const _Float16* Vhb = Vt + (size_t)hb * HEAD_DIM * KV_TOK;    // [v][g]

  // cooperative async prefetch of one 64-token K block + V block (4 b128/thread)
  auto prefetch = [&](int buf, int g0) {
    #pragma unroll
    for (int i = 0; i < 2; ++i) {
      int e   = tid + i * 256;                       // 0..511
      int row = e >> 3;                              // 0..63
      int seg = e & 7;                               // 8-half chunk within row
      async_copy_b128(lds_off(&Kbuf[buf][row][seg * 8]),
                      Khb + (size_t)(g0 + row) * HEAD_DIM + seg * 8);
      async_copy_b128(lds_off(&Vbuf[buf][row][seg * 8]),
                      Vhb + (size_t)row * KV_TOK + g0 + seg * 8);
    }
  };

  v16h aq0 = load_a_f16(Qhb, HEAD_DIM, q0, 0);
  v16h aq1 = load_a_f16(Qhb, HEAD_DIM, q0, 32);

  v8f acc[4] = {};
  float m[8], l[8];
  #pragma unroll
  for (int j = 0; j < 8; ++j) { m[j] = -3.0e38f; l[j] = 0.0f; }

  _Float16* Pw = Plds[wid];
  int rbase = (lane >> 4) << 3;
  int col0  = lane & 15;

  prefetch(0, 0);
  for (int it = 0; it < KV_TOK / 64; ++it) {
    int cur = it & 1;
    wait_async0();                                   // my copies for `cur` have landed
    __syncthreads();                                 // everyone's copies visible
    if (it + 1 < KV_TOK / 64) prefetch(cur ^ 1, (it + 1) * 64);  // overlap with compute

    const _Float16* Kl = &Kbuf[cur][0][0];
    const _Float16* Vl = &Vbuf[cur][0][0];

    // ---- scores S = NORM * Q K^T for 4 N-tiles (16 g each)
    v8f s[4];
    #pragma unroll
    for (int t = 0; t < 4; ++t) {
      v16h b0 = load_b_f16(Kl, P_LDS_LD, t * 16, 0);
      v16h b1 = load_b_f16(Kl, P_LDS_LD, t * 16, 32);
      v8f c = {};
      c = wmma_f16(aq0, b0, c);
      c = wmma_f16(aq1, b1, c);
      #pragma unroll
      for (int j = 0; j < 8; ++j) s[t][j] = c[j] * NORM_SCALE;
    }
    // ---- online softmax update
    float mn[8], al[8];
    #pragma unroll
    for (int j = 0; j < 8; ++j) {
      float bm = fmaxf(fmaxf(s[0][j], s[1][j]), fmaxf(s[2][j], s[3][j]));
      bm = red_max16(bm);
      mn[j] = fmaxf(m[j], bm);
      al[j] = __expf(m[j] - mn[j]);
      m[j]  = mn[j];
      l[j] *= al[j];
    }
    #pragma unroll
    for (int t = 0; t < 4; ++t)
      #pragma unroll
      for (int j = 0; j < 8; ++j) acc[t][j] *= al[j];

    float rs[8];
    #pragma unroll
    for (int j = 0; j < 8; ++j) rs[j] = 0.0f;
    #pragma unroll
    for (int t = 0; t < 4; ++t) {
      #pragma unroll
      for (int j = 0; j < 8; ++j) {
        float p = __expf(s[t][j] - mn[j]);
        rs[j] += p;
        Pw[(rbase + j) * P_LDS_LD + t * 16 + col0] = (_Float16)p;  // C-layout -> LDS
      }
    }
    #pragma unroll
    for (int j = 0; j < 8; ++j) l[j] += red_add16(rs[j]);

    // ---- P (A-layout from LDS) @ V (B from LDS)
    v16h ap0 = load_a_f16(Pw, P_LDS_LD, 0, 0);
    v16h ap1 = load_a_f16(Pw, P_LDS_LD, 0, 32);
    #pragma unroll
    for (int t = 0; t < 4; ++t) {
      v16h b0 = load_b_f16(Vl, P_LDS_LD, t * 16, 0);
      v16h b1 = load_b_f16(Vl, P_LDS_LD, t * 16, 32);
      acc[t] = wmma_f16(ap0, b0, acc[t]);
      acc[t] = wmma_f16(ap1, b1, acc[t]);
    }
    __syncthreads();                                 // all waves done reading `cur`
  }
  // ---- finalize: divide by l, write heads as H[b*512+q][h*64+v] f16
  float inv[8];
  #pragma unroll
  for (int j = 0; j < 8; ++j) inv[j] = 1.0f / l[j];
  #pragma unroll
  for (int t = 0; t < 4; ++t)
    #pragma unroll
    for (int j = 0; j < 8; ++j) acc[t][j] *= inv[j];

  _Float16* Hb = Hf + (size_t)b * Q_DIM * EMBED_DIM;
  #pragma unroll
  for (int t = 0; t < 4; ++t)
    store_d_f16(Hb, EMBED_DIM, q0, h * HEAD_DIM + t * 16, acc[t]);
}

// ------------------------------------------------ kernel 5: out = H(4096x512) * WoT^T -> f32
__global__ __launch_bounds__(256) void out_kernel(
    const _Float16* __restrict__ Hf, const _Float16* __restrict__ WoT,
    float* __restrict__ out) {
  int wid = threadIdx.x >> 5;
  int r0 = blockIdx.x * 128 + wid * 16;              // row in [0,4096)
  int n0 = blockIdx.y * 64;                          // embed col block
  v8f acc[4] = {};
  v16h a = load_a_f16(Hf, EMBED_DIM, r0, 0);
  v16h bb[4];
  #pragma unroll
  for (int t = 0; t < 4; ++t) bb[t] = load_b_f16(WoT, EMBED_DIM, n0 + t * 16, 0);
  for (int k0 = 0; k0 < EMBED_DIM; k0 += 32) {
    int kn = (k0 + 32) & (EMBED_DIM - 1);
    v16h a_n = load_a_f16(Hf, EMBED_DIM, r0, kn);
    v16h bb_n[4];
    #pragma unroll
    for (int t = 0; t < 4; ++t) bb_n[t] = load_b_f16(WoT, EMBED_DIM, n0 + t * 16, kn);
    #pragma unroll
    for (int t = 0; t < 4; ++t) acc[t] = wmma_f16(a, bb[t], acc[t]);
    a = a_n;
    #pragma unroll
    for (int t = 0; t < 4; ++t) bb[t] = bb_n[t];
  }
  #pragma unroll
  for (int t = 0; t < 4; ++t) store_d_f32(out, EMBED_DIM, r0, n0 + t * 16, acc[t]);
}

// ------------------------------------------------------------------ launcher
extern "C" void kernel_launch(void* const* d_in, const int* in_sizes, int n_in,
                              void* d_out, int out_size, void* d_ws, size_t ws_size,
                              hipStream_t stream) {
  (void)in_sizes; (void)n_in; (void)out_size; (void)ws_size;
  const float* q  = (const float*)d_in[0];
  const float* Wq = (const float*)d_in[1];
  const float* Wk = (const float*)d_in[2];
  const float* Wv = (const float*)d_in[3];
  const float* Wo = (const float*)d_in[4];
  float* out = (float*)d_out;

  // f16 workspace layout (~69 MB total)
  _Float16* ws  = (_Float16*)d_ws;
  _Float16* wqT = ws;                                   // 8*64*512
  _Float16* wkT = wqT + (size_t)8 * 64 * 512;
  _Float16* wvT = wkT + (size_t)8 * 64 * 512;
  _Float16* woT = wvT + (size_t)8 * 64 * 512;           // 512*512
  _Float16* Qf  = woT + (size_t)512 * 512;              // 64 * 512 * 64
  _Float16* Kf  = Qf  + (size_t)64 * Q_DIM  * HEAD_DIM; // 64 * 3584 * 64
  _Float16* Vt  = Kf  + (size_t)64 * KV_TOK * HEAD_DIM; // 64 * 64 * 3584
  _Float16* Hf  = Vt  + (size_t)64 * HEAD_DIM * KV_TOK; // 8 * 512 * 512

  prep_kernel   <<<1024,          256, 0, stream>>>(Wq, Wk, Wv, Wo, wqT, wkT, wvT, woT);
  qk_proj_kernel<<<dim3(4,  64),  256, 0, stream>>>(q, wqT, Qf, Q_DIM, 0);
  qk_proj_kernel<<<dim3(28, 64),  256, 0, stream>>>(q, wkT, Kf, KV_TOK, Q_DIM);
  v_proj_kernel <<<dim3(28, 64),  256, 0, stream>>>(q, wvT, Vt);
  attn_kernel   <<<dim3(4,  64),  256, 0, stream>>>(Qf, Kf, Vt, Hf);
  out_kernel    <<<dim3(32, 8),   256, 0, stream>>>(Hf, woT, out);
}
